// _PointnetSAModuleBase_5669356832934
// MI455X (gfx1250) — matrix operand order repaired and checked by reading, not compiled, not run on hardware
//
#include <hip/hip_runtime.h>

#define NB      8
#define NPTS    8192
#define NFEAT   64
#define NPOINT  2048
#define NSAMPLE 32
#define R2      0.04f
#define FPS_T   512

// bf16 weight blob layout inside workspace (ushort element offsets)
#define W1_OFF  0              // Wt1[64][96]
#define W2_OFF  (64 * 96)      // Wt2[64][64]
#define W3_OFF  (64 * 96 + 64 * 64)   // Wt3[128][64]
#define W_TOTAL (64 * 96 + 64 * 64 + 128 * 64)   // 18432 ushorts

typedef __attribute__((ext_vector_type(8)))  unsigned short ushort8;
typedef __attribute__((ext_vector_type(16))) unsigned short ushort16;
typedef __attribute__((ext_vector_type(16))) __bf16         bf16x16;
typedef __attribute__((ext_vector_type(8)))  float          floatx8;
typedef int int4v __attribute__((vector_size(16)));   // matches builtin proto

#if __has_builtin(__builtin_amdgcn_global_load_async_to_lds_b128)
#define HAVE_ASYNC_LDS 1
#else
#define HAVE_ASYNC_LDS 0
#endif

__device__ __forceinline__ unsigned short f2bf(float f) {
    unsigned u = __builtin_bit_cast(unsigned, f);
    u += 0x7FFFu + ((u >> 16) & 1u);   // round-to-nearest-even
    return (unsigned short)(u >> 16);
}

// copy `n` ushorts (16B-aligned chunks) global -> LDS via async DMA
__device__ __forceinline__ void copy_w_to_lds(unsigned short* dst,
                                              const unsigned short* src,
                                              int n, int tid) {
#if HAVE_ASYNC_LDS
    for (int e = tid * 8; e < n; e += 256 * 8) {
        __builtin_amdgcn_global_load_async_to_lds_b128(
            (__attribute__((address_space(1))) int4v*)(src + e),
            (__attribute__((address_space(3))) int4v*)(dst + e), 0, 0);
    }
#if __has_builtin(__builtin_amdgcn_s_wait_asynccnt)
    __builtin_amdgcn_s_wait_asynccnt(0);
#else
    asm volatile("s_wait_asynccnt 0x0" ::: "memory");
#endif
#else
    for (int e = tid * 8; e < n; e += 256 * 8)
        *(ushort8*)(dst + e) = *(const ushort8*)(src + e);
#endif
}

// ---------------------------------------------------------------------------
// One-shot weight transpose + fp32->bf16 (Wt[n][k], K zero-padded).
// ---------------------------------------------------------------------------
__global__ void prep_weights(const float* __restrict__ W1,
                             const float* __restrict__ W2,
                             const float* __restrict__ W3,
                             unsigned short* __restrict__ wbf) {
    int t = blockIdx.x * blockDim.x + threadIdx.x;
    if (t < W2_OFF) {                       // Wt1[64][96] from W1[67][64]
        int n = t / 96, k = t - n * 96;
        wbf[t] = f2bf(k < 67 ? W1[k * 64 + n] : 0.f);
    } else if (t < W3_OFF) {                // Wt2[64][64] from W2[64][64]
        int e = t - W2_OFF;
        int n = e >> 6, k = e & 63;
        wbf[t] = f2bf(W2[k * 64 + n]);
    } else if (t < W_TOTAL) {               // Wt3[128][64] from W3[64][128]
        int e = t - W3_OFF;
        int n = e >> 6, k = e & 63;
        wbf[t] = f2bf(W3[k * 128 + n]);
    }
}

// ---------------------------------------------------------------------------
// Kernel 1: furthest point sampling. One block per batch; sequential 2048-step
// argmax loop (inherently serial), points resident in registers.
// ---------------------------------------------------------------------------
__global__ void fps_kernel(const float* __restrict__ xyz,
                           int* __restrict__ fps_idx,
                           float* __restrict__ newxyz_out,
                           float* __restrict__ newxyz_ws) {
    const int b   = blockIdx.x;
    const int tid = threadIdx.x;
    const float* X = xyz + (size_t)b * NPTS * 3;

    __shared__ float smax[FPS_T];
    __shared__ int   sarg[FPS_T];

    float px[16], py[16], pz[16], mind[16];
    for (int i = 0; i < 16; i++) {
        int j = tid + i * FPS_T;
        px[i] = X[3 * j];
        py[i] = X[3 * j + 1];
        pz[i] = X[3 * j + 2];
        mind[i] = 3.4e38f;
    }

    int curidx = 0;
    for (int s = 0; s < NPOINT; s++) {
        if (tid == 0) {
            fps_idx[b * NPOINT + s] = curidx;
            size_t o = ((size_t)b * NPOINT + s) * 3;
            float cx = X[3 * curidx], cy = X[3 * curidx + 1], cz = X[3 * curidx + 2];
            newxyz_out[o] = cx; newxyz_out[o + 1] = cy; newxyz_out[o + 2] = cz;
            newxyz_ws[o]  = cx; newxyz_ws[o + 1]  = cy; newxyz_ws[o + 2]  = cz;
        }
        float cx = X[3 * curidx], cy = X[3 * curidx + 1], cz = X[3 * curidx + 2];
        float bmax = -1.f; int barg = 0;
        for (int i = 0; i < 16; i++) {
            int j = tid + i * FPS_T;
            float dx = px[i] - cx, dy = py[i] - cy, dz = pz[i] - cz;
            float d = dx * dx + dy * dy + dz * dz;
            if (d < mind[i]) mind[i] = d;
            if (mind[i] > bmax) { bmax = mind[i]; barg = j; }
        }
        smax[tid] = bmax; sarg[tid] = barg;
        __syncthreads();
        for (int off = FPS_T / 2; off > 0; off >>= 1) {
            if (tid < off) {
                float mo = smax[tid + off]; int ao = sarg[tid + off];
                if (mo > smax[tid] || (mo == smax[tid] && ao < sarg[tid])) {
                    smax[tid] = mo; sarg[tid] = ao;
                }
            }
            __syncthreads();
        }
        curidx = sarg[0];
        __syncthreads();
    }
}

// ---------------------------------------------------------------------------
// Kernel 2: ball query. One thread per center; collect first NSAMPLE indices
// (ascending order == jnp.sort of masked indices), pad with first hit.
// ---------------------------------------------------------------------------
__global__ void ballquery_kernel(const float* __restrict__ xyz,
                                 const float* __restrict__ newxyz,
                                 int* __restrict__ ball_idx) {
    int g = blockIdx.x * blockDim.x + threadIdx.x;
    if (g >= NB * NPOINT) return;
    int b = g / NPOINT;
    const float* X = xyz + (size_t)b * NPTS * 3;
    float cx = newxyz[3 * g], cy = newxyz[3 * g + 1], cz = newxyz[3 * g + 2];
    int* out = ball_idx + (size_t)g * NSAMPLE;
    int cnt = 0, first = -1;
    for (int j = 0; j < NPTS && cnt < NSAMPLE; j++) {
        __builtin_prefetch(&X[3 * j + 96], 0, 0);
        float dx = X[3 * j] - cx, dy = X[3 * j + 1] - cy, dz = X[3 * j + 2] - cz;
        float d2 = dx * dx + dy * dy + dz * dz;
        if (d2 < R2) {
            if (first < 0) first = j;
            out[cnt++] = j;
        }
    }
    if (first < 0) first = 0;
    for (; cnt < NSAMPLE; cnt++) out[cnt] = first;
}

// ---------------------------------------------------------------------------
// Fused gather + 3-layer MLP (bf16 WMMA) + max-pool.
// Block = 256 threads (8 waves), owns 4 centers x 32 samples = 128 rows.
//   A fragment (16-bit, 16x32): lane<16 -> K {k0..k0+7, k0+16..k0+23},
//                               lane>=16 -> K {k0+8.., k0+24..}; row = lane&15.
//   B fragment (16-bit, 32x16): lane holds col=lane&15, same K split as A
//                               (k0+8*(lane>=16) + {0..7, 16..23}); weights stored Wt[n][k].
//   C (f32 16x16): vgpr r -> row r + 8*(lane>=16), col = lane&15.
// nt = t%NT is per-wave constant, so B fragments stay register-resident.
// ---------------------------------------------------------------------------
template <int NT, int KT, bool POOL>
__device__ __forceinline__ void mlp_layer(const unsigned short* A, int lda,
                                          const unsigned short* Wt, int ldw,
                                          const float* __restrict__ bias,
                                          unsigned short* Out, int ldo,
                                          float* pool, int tid) {
    const int MT_TILES = 8;                 // 128 rows / 16
    int wave = tid >> 5, lane = tid & 31;
    int half = lane >> 4;                   // 0 | 1
    int l16  = lane & 15;
    for (int t = wave; t < MT_TILES * NT; t += 8) {
        int mt = t / NT, nt = t % NT;
        int arow = mt * 16 + l16;
        int bcol = nt * 16 + l16;
        floatx8 c = {0.f, 0.f, 0.f, 0.f, 0.f, 0.f, 0.f, 0.f};
#pragma unroll
        for (int kb = 0; kb < KT; kb++) {
            int k0 = kb * 32;
            const unsigned short* pa = A + arow * lda + k0 + half * 8;
            ushort8 a0 = *(const ushort8*)pa;
            ushort8 a1 = *(const ushort8*)(pa + 16);
            const unsigned short* pb = Wt + bcol * ldw + k0 + half * 8;
            ushort8 b0 = *(const ushort8*)pb;
            ushort8 b1 = *(const ushort8*)(pb + 16);
            ushort16 av, bv;
#pragma unroll
            for (int i = 0; i < 8; i++) {
                av[i] = a0[i]; av[i + 8] = a1[i];
                bv[i] = b0[i]; bv[i + 8] = b1[i];
            }
            bf16x16 af = __builtin_bit_cast(bf16x16, av);
            bf16x16 bf = __builtin_bit_cast(bf16x16, bv);
            c = __builtin_amdgcn_wmma_f32_16x16x32_bf16(
                    false, af, false, bf, (short)0, c, false, false);
        }
        float bb = bias[bcol];
        int rowbase = mt * 16 + half * 8;
#pragma unroll
        for (int r = 0; r < 8; r++) {
            float v = c[r] + bb;
            v = v > 0.f ? v : 0.f;
            if (POOL) {
                int row = rowbase + r;
                int ctr = row >> 5;         // center within block
                atomicMax((int*)&pool[ctr * 128 + bcol], __float_as_int(v));
            } else {
                Out[(rowbase + r) * ldo + bcol] = f2bf(v);
            }
        }
    }
}

__global__ void __launch_bounds__(256)
sa_mlp_kernel(const float* __restrict__ xyz, const float* __restrict__ feat,
              const float* __restrict__ newxyz, const int* __restrict__ ball_idx,
              const unsigned short* __restrict__ wbf,
              const float* __restrict__ b1, const float* __restrict__ b2,
              const float* __restrict__ b3,
              float* __restrict__ out_feat) {
    __shared__ __align__(16) unsigned short sA[128 * 96];   // 24 KB  A0 / A2
    __shared__ __align__(16) unsigned short sB[128 * 64];   // 16 KB  A1
    __shared__ __align__(16) unsigned short sW[128 * 64];   // 16 KB  Wt (reused)
    __shared__ float sPool[4 * 128];                        //  2 KB
    __shared__ int   sIdx[128];
    __shared__ float sCtr[12];

    const int blk = blockIdx.x;
    const int b   = blk / (NPOINT / 4);
    const int s0  = (blk % (NPOINT / 4)) * 4;
    const int tid = threadIdx.x;

    if (tid < 128)
        sIdx[tid] = ball_idx[((size_t)(b * NPOINT + s0 + (tid >> 5))) * NSAMPLE + (tid & 31)];
    if (tid < 12)
        sCtr[tid] = newxyz[((size_t)b * NPOINT + s0) * 3 + tid];
    for (int i = tid; i < 4 * 128; i += 256) sPool[i] = 0.f;

    // ---- async-copy Wt1 (completed before the gather below) ----
    copy_w_to_lds(sW, wbf + W1_OFF, 64 * 96, tid);
    __syncthreads();

    // ---- gather: A0[128 x 96] = [rel_xyz(3) | feat(64) | zero-pad(29)] ----
    const float* X = xyz  + (size_t)b * NPTS * 3;
    const float* F = feat + (size_t)b * NPTS * NFEAT;
    for (int e = tid; e < 128 * 96; e += 256) {
        int row = e / 96, col = e - row * 96;
        int j = sIdx[row];
        float v;
        if (col < 3)       v = X[3 * j + col] - sCtr[(row >> 5) * 3 + col];
        else if (col < 67) v = F[(size_t)j * NFEAT + (col - 3)];
        else               v = 0.f;
        sA[row * 96 + col] = f2bf(v);
    }
    __syncthreads();
    mlp_layer<4, 3, false>(sA, 96, sW, 96, b1, sB, 64, nullptr, tid);   // 128x67 @ 67x64
    __syncthreads();

    copy_w_to_lds(sW, wbf + W2_OFF, 64 * 64, tid);
    __syncthreads();
    mlp_layer<4, 2, false>(sB, 64, sW, 64, b2, sA, 64, nullptr, tid);   // 128x64 @ 64x64
    __syncthreads();

    copy_w_to_lds(sW, wbf + W3_OFF, 128 * 64, tid);
    __syncthreads();
    mlp_layer<8, 2, true>(sA, 64, sW, 64, b3, nullptr, 0, sPool, tid);  // 128x64 @ 64x128 + pool
    __syncthreads();

    // ---- write new_features (B, 128, NPOINT) ----
    for (int e = tid; e < 4 * 128; e += 256) {
        int c = e >> 7, d = e & 127;
        out_feat[((size_t)b * 128 + d) * NPOINT + (s0 + c)] = sPool[c * 128 + d];
    }
}

// ---------------------------------------------------------------------------
extern "C" void kernel_launch(void* const* d_in, const int* in_sizes, int n_in,
                              void* d_out, int out_size, void* d_ws, size_t ws_size,
                              hipStream_t stream) {
    (void)in_sizes; (void)n_in; (void)out_size; (void)ws_size;
    const float* xyz  = (const float*)d_in[0];
    const float* feat = (const float*)d_in[1];
    const float* W1   = (const float*)d_in[2];
    const float* b1   = (const float*)d_in[3];
    const float* W2   = (const float*)d_in[4];
    const float* b2   = (const float*)d_in[5];
    const float* W3   = (const float*)d_in[6];
    const float* b3   = (const float*)d_in[7];

    float* out      = (float*)d_out;
    float* out_xyz  = out;                               // B*NPOINT*3
    float* out_feat = out + (size_t)NB * NPOINT * 3;     // B*128*NPOINT

    char*  ws        = (char*)d_ws;
    int*   fps_idx   = (int*)ws;                                   // 64 KB
    float* newxyz_ws = (float*)(ws + 65536);                       // 192 KB
    int*   ball_idx  = (int*)(ws + 65536 + 196608);                // 2 MB
    unsigned short* wbf = (unsigned short*)(ws + 65536 + 196608 + 2097152); // 36 KB

    prep_weights<<<(W_TOTAL + 255) / 256, 256, 0, stream>>>(W1, W2, W3, wbf);
    fps_kernel<<<NB, FPS_T, 0, stream>>>(xyz, fps_idx, out_xyz, newxyz_ws);
    ballquery_kernel<<<(NB * NPOINT + 255) / 256, 256, 0, stream>>>(xyz, newxyz_ws, ball_idx);
    sa_mlp_kernel<<<NB * (NPOINT / 4), 256, 0, stream>>>(
        xyz, feat, newxyz_ws, ball_idx, wbf, b1, b2, b3, out_feat);
}
